// DifferentialMultiHeadAttention_35802847380247
// MI455X (gfx1250) — matrix operand order repaired and checked by reading, not compiled, not run on hardware
//
#include <hip/hip_runtime.h>
#include <hip/hip_bf16.h>

typedef __bf16 bf16;
typedef __attribute__((ext_vector_type(16))) __bf16 v16bf;
typedef __attribute__((ext_vector_type(8)))  __bf16 v8bf;
typedef __attribute__((ext_vector_type(8)))  float   v8f;

#define BB 2
#define TT 2048
#define CC 1024
#define HH 16
#define DD 64
#define MM (BB * TT)          // 4096 rows of the flattened [B*T, C] activations
#define LAM_INIT 0.2f
#define RMS_EPS 1e-8f
#define LOG2E 1.44269504088896340736f

static __device__ __forceinline__ v8f vzero8() {
  v8f z;
#pragma unroll
  for (int i = 0; i < 8; ++i) z[i] = 0.0f;
  return z;
}

// A-operand (16 rows x 32 k) of v_wmma_f32_16x16x32_bf16, loaded from a
// row-major bf16 matrix. Lanes 0-15: row = lane, k = {0..7, 16..23};
// lanes 16-31: row = lane-16, k = {8..15, 24..31}. Two 16B loads per lane.
static __device__ __forceinline__ v16bf ld_A(const bf16* __restrict__ base, int ld,
                                             int m0, int k0, int lane) {
  int l = lane & 15, h = (lane >> 4) & 1;
  const bf16* p = base + (size_t)(m0 + l) * ld + (k0 + h * 8);
  v8bf lo = *reinterpret_cast<const v8bf*>(p);
  v8bf hi = *reinterpret_cast<const v8bf*>(p + 16);
  return __builtin_shufflevector(lo, hi, 0, 1, 2, 3, 4, 5, 6, 7,
                                 8, 9, 10, 11, 12, 13, 14, 15);
}

// B-operand (32 k x 16 cols) loaded from row-major B^T (i.e. [col][k]).
// Lanes 0-15: col = lane, k = k0+{0..7, 16..23}; lanes 16-31: col = lane-16, k = k0+{8..15, 24..31}.
static __device__ __forceinline__ v16bf ld_B(const bf16* __restrict__ base, int ld,
                                             int n0, int k0, int lane) {
  int l = lane & 15, h = (lane >> 4) & 1;
  const bf16* p = base + (size_t)(n0 + l) * ld + (k0 + h * 8);
  v8bf lo = *reinterpret_cast<const v8bf*>(p);
  v8bf hi = *reinterpret_cast<const v8bf*>(p + 16);
  return __builtin_shufflevector(lo, hi, 0, 1, 2, 3, 4, 5, 6, 7,
                                 8, 9, 10, 11, 12, 13, 14, 15);
}

static __device__ __forceinline__ v8f wmma_bf16(v16bf a, v16bf b, v8f c) {
  return __builtin_amdgcn_wmma_f32_16x16x32_bf16(false, a, false, b, (short)0, c,
                                                 false, false);
}

// ---------------------------------------------------------------- converts
__global__ void cvt_f32_bf16(const float* __restrict__ s, bf16* __restrict__ d, int n) {
  int i = blockIdx.x * blockDim.x + threadIdx.x;
  if (i < n) d[i] = (bf16)s[i];
}

// Wt[n*K + k] = (bf16) W[k*N + n]   (store-coalesced transpose+convert)
__global__ void transpose_cvt(const float* __restrict__ W, bf16* __restrict__ Wt,
                              int K, int N) {
  int i = blockIdx.x * blockDim.x + threadIdx.x;
  if (i < K * N) {
    int n = i / K, k = i - n * K;
    Wt[i] = (bf16)W[(size_t)k * N + n];
  }
}

// ---------------------------------------------------------------- lambda
__global__ void lambda_kernel(const float* __restrict__ lq1, const float* __restrict__ lk1,
                              const float* __restrict__ lq2, const float* __restrict__ lk2,
                              float* __restrict__ lam) {
  int l = threadIdx.x;  // 32 threads, D = 64
  float s1 = lq1[l] * lk1[l] + lq1[l + 32] * lk1[l + 32];
  float s2 = lq2[l] * lk2[l] + lq2[l + 32] * lk2[l + 32];
#pragma unroll
  for (int off = 16; off >= 1; off >>= 1) {
    s1 += __shfl_xor(s1, off, 32);
    s2 += __shfl_xor(s2, off, 32);
  }
  if (l == 0) *lam = __expf(s1) - __expf(s2) + LAM_INIT;
}

// ---------------------------------------------------------------- GEMM
// Y = A[M x K] @ Bt^T, A bf16 row-major, Bt bf16 stored [N][K].
// One wave computes a 32(M) x 64(N) tile: 2 A-tiles, 4 B-tiles, 8 WMMAs
// per k-step (B reused across both M-subtiles -> ~11 MACs/byte from L2).
// mode 0: bf16 out, head-major [B,H,T,D]
// mode 1: bf16 out, V-transposed [B,H,D,T]
// mode 2: f32 out, row-major [M,N]
__global__ __launch_bounds__(32) void gemm_wmma(const bf16* __restrict__ A,
                                                const bf16* __restrict__ Bt,
                                                void* __restrict__ out, int mode) {
  int lane = threadIdx.x;
  int n0 = blockIdx.x * 64;
  int m0 = blockIdx.y * 32;
  v8f acc[2][4];
#pragma unroll
  for (int i = 0; i < 2; ++i)
#pragma unroll
    for (int j = 0; j < 4; ++j) acc[i][j] = vzero8();

  for (int k0 = 0; k0 < CC; k0 += 32) {
    v16bf a0 = ld_A(A, CC, m0, k0, lane);
    v16bf a1 = ld_A(A, CC, m0 + 16, k0, lane);
#pragma unroll
    for (int j = 0; j < 4; ++j) {
      v16bf b = ld_B(Bt, CC, n0 + 16 * j, k0, lane);
      acc[0][j] = wmma_bf16(a0, b, acc[0][j]);
      acc[1][j] = wmma_bf16(a1, b, acc[1][j]);
    }
  }

  int l = lane & 15, h = (lane >> 4) & 1;
#pragma unroll
  for (int i = 0; i < 2; ++i) {
#pragma unroll
    for (int j = 0; j < 4; ++j) {
#pragma unroll
      for (int r = 0; r < 8; ++r) {
        float v = acc[i][j][r];
        int row = m0 + 16 * i + h * 8 + r;  // m index in [0, M)
        int col = n0 + 16 * j + l;          // n index in [0, N)
        if (mode == 2) {
          ((float*)out)[(size_t)row * CC + col] = v;
        } else {
          int bidx = row >> 11, t = row & (TT - 1);
          int hh = col >> 6, dd = col & 63;
          size_t idx = (mode == 0)
              ? ((size_t)(bidx * HH + hh) * TT + t) * DD + dd
              : ((size_t)(bidx * HH + hh) * DD + dd) * TT + t;
          ((bf16*)out)[idx] = (bf16)v;
        }
      }
    }
  }
}

// ---------------------------------------------------------------- attention
// One wave per (b*H+h, 16-query tile). Q/K head-major [BH][T][D], Vt [BH][D][T].
// Softmax is computed entirely in the base-2 domain (log2e folded into the
// 1/sqrt(D) scale) so every exponential is a single v_exp_f32.
__global__ __launch_bounds__(32) void diff_attn_kernel(
    const bf16* __restrict__ Q1, const bf16* __restrict__ Q2,
    const bf16* __restrict__ K1, const bf16* __restrict__ K2,
    const bf16* __restrict__ Vt, const float* __restrict__ lam_p,
    const float* __restrict__ norm_w, float* __restrict__ diff_out,
    bf16* __restrict__ Obuf) {
  __shared__ __align__(16) bf16 Pst[16][32];

  int lane = threadIdx.x;
  int bh = blockIdx.y;
  int q0 = blockIdx.x * 16;
  int l = lane & 15, hf = (lane >> 4) & 1;

  const bf16* q1b = Q1 + (size_t)bh * TT * DD;
  const bf16* q2b = Q2 + (size_t)bh * TT * DD;
  const bf16* k1b = K1 + (size_t)bh * TT * DD;
  const bf16* k2b = K2 + (size_t)bh * TT * DD;
  const bf16* vtb = Vt + (size_t)bh * DD * TT;

  // Q tiles live in registers for the whole kernel (A-operand layout).
  v16bf a1lo = ld_A(q1b, DD, q0, 0, lane);
  v16bf a1hi = ld_A(q1b, DD, q0, 32, lane);
  v16bf a2lo = ld_A(q2b, DD, q0, 0, lane);
  v16bf a2hi = ld_A(q2b, DD, q0, 32, lane);

  const float sc2 = 0.125f * LOG2E;  // D^-0.5 * log2(e)
  float m1[8], s1s[8], m2[8], s2s[8];
#pragma unroll
  for (int r = 0; r < 8; ++r) { m1[r] = -3.0e38f; m2[r] = -3.0e38f; s1s[r] = 0.f; s2s[r] = 0.f; }

  // ---- pass 1: online max / sum-exp2 per row for both score matrices
  for (int k0 = 0; k0 < TT; k0 += 16) {
    if (k0 + 32 < TT) {  // prefetch K tiles 2 iterations ahead
      __builtin_prefetch((const void*)(k1b + (size_t)(k0 + 32) * DD), 0, 1);
      __builtin_prefetch((const void*)(k2b + (size_t)(k0 + 32) * DD), 0, 1);
    }
    v16bf bk1 = ld_B(k1b, DD, k0, 0, lane);
    v16bf bk1h = ld_B(k1b, DD, k0, 32, lane);
    v8f s1 = vzero8();
    s1 = wmma_bf16(a1lo, bk1, s1);
    s1 = wmma_bf16(a1hi, bk1h, s1);
    v16bf bk2 = ld_B(k2b, DD, k0, 0, lane);
    v16bf bk2h = ld_B(k2b, DD, k0, 32, lane);
    v8f s2 = vzero8();
    s2 = wmma_bf16(a2lo, bk2, s2);
    s2 = wmma_bf16(a2hi, bk2h, s2);
#pragma unroll
    for (int r = 0; r < 8; ++r) {
      float v1 = s1[r] * sc2;
      float mn1 = fmaxf(m1[r], v1);
      s1s[r] = s1s[r] * __builtin_exp2f(m1[r] - mn1) + __builtin_exp2f(v1 - mn1);
      m1[r] = mn1;
      float v2 = s2[r] * sc2;
      float mn2 = fmaxf(m2[r], v2);
      s2s[r] = s2s[r] * __builtin_exp2f(m2[r] - mn2) + __builtin_exp2f(v2 - mn2);
      m2[r] = mn2;
    }
  }
  // cross-lane merge within each 16-lane half (lanes share the same 8 rows)
#pragma unroll
  for (int off = 8; off >= 1; off >>= 1) {
#pragma unroll
    for (int r = 0; r < 8; ++r) {
      float mo = __shfl_xor(m1[r], off, 32);
      float so = __shfl_xor(s1s[r], off, 32);
      float mn = fmaxf(m1[r], mo);
      s1s[r] = s1s[r] * __builtin_exp2f(m1[r] - mn) + so * __builtin_exp2f(mo - mn);
      m1[r] = mn;
      mo = __shfl_xor(m2[r], off, 32);
      so = __shfl_xor(s2s[r], off, 32);
      mn = fmaxf(m2[r], mo);
      s2s[r] = s2s[r] * __builtin_exp2f(m2[r] - mn) + so * __builtin_exp2f(mo - mn);
      m2[r] = mn;
    }
  }
  float il1[8], il2[8];
#pragma unroll
  for (int r = 0; r < 8; ++r) { il1[r] = 1.0f / s1s[r]; il2[r] = 1.0f / s2s[r]; }

  float lam = *lam_p;

  // ---- pass 2: recompute S, emit diff_attn, accumulate O = diff @ V
  v8f o[4];
#pragma unroll
  for (int j = 0; j < 4; ++j) o[j] = vzero8();

  for (int k0 = 0; k0 < TT; k0 += 32) {
    if (k0 + 64 < TT) {
      __builtin_prefetch((const void*)(k1b + (size_t)(k0 + 64) * DD), 0, 1);
      __builtin_prefetch((const void*)(k2b + (size_t)(k0 + 64) * DD), 0, 1);
    }
#pragma unroll
    for (int ks = 0; ks < 2; ++ks) {
      int kk = k0 + ks * 16;
      v16bf bk1 = ld_B(k1b, DD, kk, 0, lane);
      v16bf bk1h = ld_B(k1b, DD, kk, 32, lane);
      v8f s1 = vzero8();
      s1 = wmma_bf16(a1lo, bk1, s1);
      s1 = wmma_bf16(a1hi, bk1h, s1);
      v16bf bk2 = ld_B(k2b, DD, kk, 0, lane);
      v16bf bk2h = ld_B(k2b, DD, kk, 32, lane);
      v8f s2 = vzero8();
      s2 = wmma_bf16(a2lo, bk2, s2);
      s2 = wmma_bf16(a2hi, bk2h, s2);
#pragma unroll
      for (int r = 0; r < 8; ++r) {
        float p1 = __builtin_exp2f(s1[r] * sc2 - m1[r]) * il1[r];
        float p2 = __builtin_exp2f(s2[r] * sc2 - m2[r]) * il2[r];
        float dv = p1 - lam * p2;
        int row = hf * 8 + r;
        // streaming 537 MB output: non-temporal store
        __builtin_nontemporal_store(
            dv, &diff_out[((size_t)bh * TT + q0 + row) * TT + kk + l]);
        Pst[row][ks * 16 + l] = (bf16)dv;
      }
    }
    __syncthreads();
    // re-read the 16x32 diff tile in A-operand layout
    const bf16* pp = &Pst[l][hf * 8];
    v8bf plo = *reinterpret_cast<const v8bf*>(pp);
    v8bf phi = *reinterpret_cast<const v8bf*>(pp + 16);
    v16bf pa = __builtin_shufflevector(plo, phi, 0, 1, 2, 3, 4, 5, 6, 7,
                                       8, 9, 10, 11, 12, 13, 14, 15);
#pragma unroll
    for (int j = 0; j < 4; ++j) {
      v16bf vb = ld_B(vtb, TT, j * 16, k0, lane);  // cols = d, k = key index
      o[j] = wmma_bf16(pa, vb, o[j]);
    }
    __syncthreads();
  }

  // ---- epilogue: per-row RMSNorm over D=64, scale by (1 - lambda_init)
  float w[4];
#pragma unroll
  for (int j = 0; j < 4; ++j) w[j] = norm_w[16 * j + l];
  int bidx = bh / HH, hidx = bh % HH;
#pragma unroll
  for (int r = 0; r < 8; ++r) {
    float ss = 0.f;
#pragma unroll
    for (int j = 0; j < 4; ++j) ss += o[j][r] * o[j][r];
#pragma unroll
    for (int off = 8; off >= 1; off >>= 1) ss += __shfl_xor(ss, off, 32);
    float scale = (1.0f - LAM_INIT) * rsqrtf(ss * (1.0f / 64.0f) + RMS_EPS);
    int rowg = bidx * TT + q0 + hf * 8 + r;  // row of [B*T, C]
#pragma unroll
    for (int j = 0; j < 4; ++j) {
      float val = o[j][r] * scale * w[j];
      Obuf[(size_t)rowg * CC + hidx * 64 + 16 * j + l] = (bf16)val;
    }
  }
}

// ---------------------------------------------------------------- launch
extern "C" void kernel_launch(void* const* d_in, const int* in_sizes, int n_in,
                              void* d_out, int out_size, void* d_ws, size_t ws_size,
                              hipStream_t stream) {
  const float* x   = (const float*)d_in[0];
  const float* Ws[6] = {(const float*)d_in[1], (const float*)d_in[2],
                        (const float*)d_in[3], (const float*)d_in[4],
                        (const float*)d_in[5], (const float*)d_in[6]};
  const float* lq1 = (const float*)d_in[7];
  const float* lk1 = (const float*)d_in[8];
  const float* lq2 = (const float*)d_in[9];
  const float* lk2 = (const float*)d_in[10];
  const float* nw  = (const float*)d_in[11];

  char* ws = (char*)d_ws;
  size_t off = 0;
  float* lam = (float*)ws;                 off += 256;
  bf16* xbf  = (bf16*)(ws + off);          off += (size_t)MM * CC * 2;
  bf16* wt[6];
  for (int i = 0; i < 6; ++i) { wt[i] = (bf16*)(ws + off); off += (size_t)CC * CC * 2; }
  bf16* Q1   = (bf16*)(ws + off);          off += (size_t)MM * CC * 2;
  bf16* Q2   = (bf16*)(ws + off);          off += (size_t)MM * CC * 2;
  bf16* K1h  = (bf16*)(ws + off);          off += (size_t)MM * CC * 2;
  bf16* K2h  = (bf16*)(ws + off);          off += (size_t)MM * CC * 2;
  bf16* Vt   = (bf16*)(ws + off);          off += (size_t)MM * CC * 2;
  bf16* Obuf = (bf16*)(ws + off);          off += (size_t)MM * CC * 2;

  int nx = MM * CC;
  cvt_f32_bf16<<<(nx + 255) / 256, 256, 0, stream>>>(x, xbf, nx);
  int nw_elems = CC * CC;
  for (int i = 0; i < 6; ++i)
    transpose_cvt<<<(nw_elems + 255) / 256, 256, 0, stream>>>(Ws[i], wt[i], CC, CC);
  lambda_kernel<<<1, 32, 0, stream>>>(lq1, lk1, lq2, lk2, lam);

  dim3 gg(CC / 64, MM / 32);
  gemm_wmma<<<gg, 32, 0, stream>>>(xbf, wt[0], Q1, 0);
  gemm_wmma<<<gg, 32, 0, stream>>>(xbf, wt[1], Q2, 0);
  gemm_wmma<<<gg, 32, 0, stream>>>(xbf, wt[2], K1h, 0);
  gemm_wmma<<<gg, 32, 0, stream>>>(xbf, wt[3], K2h, 0);
  gemm_wmma<<<gg, 32, 0, stream>>>(xbf, wt[4], Vt, 1);

  float* out_p = (float*)d_out;
  float* diff_p = out_p + (size_t)MM * CC;
  diff_attn_kernel<<<dim3(TT / 16, BB * HH), 32, 0, stream>>>(
      Q1, Q2, K1h, K2h, Vt, lam, nw, diff_p, Obuf);

  gemm_wmma<<<gg, 32, 0, stream>>>(Obuf, wt[5], d_out, 2);
}
